// NewSegaMemTransformerLM_10934986736217
// MI455X (gfx1250) — matrix operand order repaired and checked by reading, not compiled, not run on hardware
//
#include <hip/hip_runtime.h>
#include <hip/hip_bf16.h>
#include <stdint.h>

// ---------------- problem constants ----------------
#define QLEN 1024
#define MLEN 1024
#define KLEN 2048
#define BSZ  4
#define DMODEL 512
#define NHEAD 8
#define DHEAD 64
#define DFF  2048
#define NLAYER 3

typedef __attribute__((ext_vector_type(16))) __bf16 v16bf;
typedef __attribute__((ext_vector_type(8)))  float  v8f;
typedef unsigned short u16;

// A/B operand fragment: 16 bf16 values per lane = 32 bytes = two b128 loads
union ABFrag { v16bf v; uint4 q[2]; };

static __device__ __forceinline__ u16 f32_to_bf16(float f) {
  union { float f; unsigned int u; } v; v.f = f;
  unsigned int r = v.u + 0x7FFFu + ((v.u >> 16) & 1u);   // RNE
  return (u16)(r >> 16);
}

// one 16B-per-lane async DMA chunk: global -> LDS (tracked by ASYNCcnt)
static __device__ __forceinline__ void async_ld16(unsigned lds_off, const void* g) {
  asm volatile("global_load_async_to_lds_b128 %0, %1, off"
               :: "v"(lds_off), "v"((unsigned long long)(uintptr_t)g) : "memory");
}

// ---------------- embedding + positional embedding ----------------
// grid: QLEN*BSZ blocks (row = i*BSZ+b), 256 threads, 2 elems/thread
__global__ void embed_kernel(const int* __restrict__ inp, const float* __restrict__ emb,
                             float* __restrict__ h) {
  int row = blockIdx.x;
  int i = row >> 2;
  int tok = inp[row];
  float t = 1023.0f - (float)i;      // pos_seq value at position (mlen+i)
  for (int e = 0; e < 2; ++e) {
    int d = threadIdx.x + e * 256;
    float v = emb[(size_t)tok * DMODEL + d] * 22.62741699796952f;  // sqrt(512)
    int j = (d < 256) ? d : (d - 256);
    float freq = __expf(-9.210340371976184f * ((float)(2 * j) * (1.0f / 512.0f)));
    float ang = t * freq;
    v += (d < 256) ? __sinf(ang) : __cosf(ang);
    h[(size_t)row * DMODEL + d] = v;
  }
}

// ---------------- f32 -> bf16 convert ----------------
__global__ void f2bf_kernel(const float* __restrict__ src, u16* __restrict__ dst, int n) {
  int i = blockIdx.x * 256 + threadIdx.x;
  if (i < n) dst[i] = f32_to_bf16(src[i]);
}

// ---------------- weight transpose + convert: W[K,N] f32 -> Wt[N,K] bf16 ----------------
__global__ void transp_kernel(const float* __restrict__ W, u16* __restrict__ Wt, int K, int N) {
  int idx = blockIdx.x * 256 + threadIdx.x;
  if (idx >= K * N) return;
  int k = idx / N, n = idx % N;
  Wt[(size_t)n * K + k] = f32_to_bf16(W[idx]);
}

// ---------------- v transpose: kv_bf[(j*BSZ+b),1024] -> vT[b][h][d][j] ----------------
__global__ void vtrans_kernel(const u16* __restrict__ kv_bf, u16* __restrict__ vT) {
  int idx = blockIdx.x * 256 + threadIdx.x;   // 4*8*64*2048 = 4194304 total
  int j = idx & (KLEN - 1);
  int d = (idx >> 11) & 63;
  int hh = (idx >> 17) & 7;
  int b = idx >> 20;
  vT[idx] = kv_bf[((size_t)j * BSZ + b) * (2 * NHEAD * DHEAD) + (NHEAD * DHEAD) + hh * DHEAD + d];
}

// ---------------- WMMA GEMM: C[M,N] = A[M,K](bf16) x Bt[N,K](bf16)^T (+bias)(+relu) ----------
// block: 128 threads = 4 waves in 2x2; wave tile 32x64; block tile 64x128.
// Double-buffered software pipeline: k-tile (A 64x32 + B 128x32) is DMA'd into
// LDS with GLOBAL_LOAD_ASYNC_TO_LDS_B128 while the previous tile feeds 8 WMMAs;
// in-order ASYNCcnt lets `s_wait_asynccnt 6` drain exactly the older tile.
// grid (N/128, M/64)
__global__ void wmma_gemm_bt(const u16* __restrict__ A, int lda,
                             const u16* __restrict__ Bt, int ldb,
                             float* __restrict__ Cf, u16* __restrict__ Cb, int ldc,
                             const float* __restrict__ bias,
                             int K, int relu) {
  __shared__ __align__(16) u16 tileA[2][64][32];    // [buf][m in block][k half]
  __shared__ __align__(16) u16 tileB[2][128][32];   // [buf][n in block][k half]
  const int tid = threadIdx.x;
  const int lane = tid & 31;
  const int wave = tid >> 5;
  const int wm = wave >> 1, wn = wave & 1;
  const int mBase = blockIdx.y * 64;
  const int nBase = blockIdx.x * 128;
  const int am = lane & 15, ah = lane >> 4;

  // uniform-trip staging: each thread issues exactly 6 async 16B chunks per k-tile
  const int sr = tid >> 2;          // 0..31
  const int ssub = tid & 3;         // 0..3 (16B sub-chunk of a 64B row)
  auto stage = [&](int k, int buf) {
    async_ld16((unsigned)(uintptr_t)&tileA[buf][sr][ssub * 8],
               A + (size_t)(mBase + sr) * lda + k + ssub * 8);
    async_ld16((unsigned)(uintptr_t)&tileA[buf][sr + 32][ssub * 8],
               A + (size_t)(mBase + sr + 32) * lda + k + ssub * 8);
#pragma unroll
    for (int it = 0; it < 4; ++it)
      async_ld16((unsigned)(uintptr_t)&tileB[buf][sr + it * 32][ssub * 8],
                 Bt + (size_t)(nBase + sr + it * 32) * ldb + k + ssub * 8);
  };

  v8f acc[2][4];
  for (int i = 0; i < 2; ++i)
    for (int t = 0; t < 4; ++t)
      for (int r = 0; r < 8; ++r) acc[i][t][r] = 0.0f;

  const int nk = K >> 5;
  stage(0, 0);                                    // prologue: tile 0 -> buf 0
  for (int kk = 0; kk < nk; ++kk) {
    const int buf = kk & 1;
    if (kk + 1 < nk) {
      stage((kk + 1) << 5, buf ^ 1);              // prefetch next tile
      asm volatile("s_wait_asynccnt 0x6" ::: "memory");   // drain older tile only
    } else {
      asm volatile("s_wait_asynccnt 0x0" ::: "memory");
    }
    __syncthreads();

    // ---- fragments from LDS ----
    ABFrag a[2], b[4];
    for (int i = 0; i < 2; ++i) {
      const u16* lp = &tileA[buf][wm * 32 + i * 16 + am][0];
      a[i].q[0] = *(const uint4*)(lp + ah * 8);          // K = half*8 + j
      a[i].q[1] = *(const uint4*)(lp + 16 + ah * 8);     // K = 16 + half*8 + j
    }
    for (int t = 0; t < 4; ++t) {
      const u16* lp = &tileB[buf][wn * 64 + t * 16 + am][0];
      b[t].q[0] = *(const uint4*)(lp + ah * 8);          // K = half*8 + j
      b[t].q[1] = *(const uint4*)(lp + 16 + ah * 8);     // K = 16 + half*8 + j
    }
    // ---- 8 WMMAs per k-step ----
    for (int i = 0; i < 2; ++i)
      for (int t = 0; t < 4; ++t)
        acc[i][t] = __builtin_amdgcn_wmma_f32_16x16x32_bf16(false, a[i].v, false, b[t].v,
                                                            (short)0, acc[i][t], false, false);
    __syncthreads();   // all fragment reads done before this buffer is re-staged
  }

  const int cn = lane & 15, chalf = lane >> 4;   // C/D: M = r + 8*half, N = lane&15
  for (int t = 0; t < 4; ++t) {
    int col = nBase + wn * 64 + t * 16 + cn;
    float bv = bias ? bias[col] : 0.0f;
    for (int i = 0; i < 2; ++i)
      for (int r = 0; r < 8; ++r) {
        int row = mBase + wm * 32 + i * 16 + chalf * 8 + r;
        float v = acc[i][t][r] + bv;
        if (relu) v = fmaxf(v, 0.0f);
        if (Cf) Cf[(size_t)row * ldc + col] = v;
        if (Cb) Cb[(size_t)row * ldc + col] = f32_to_bf16(v);
      }
  }
}

// ---------------- flash attention ----------------
// grid: (QLEN/32, BSZ*NHEAD); block 64 threads = 2 waves (16 query rows each)
__global__ void flash_attn(const u16* __restrict__ q_bf, const u16* __restrict__ kv_bf,
                           const u16* __restrict__ vT, u16* __restrict__ vec_bf) {
  __shared__ __align__(16) u16 pls[2][16][32];   // per-wave P tile (C-layout -> A-layout)
  int b = blockIdx.y >> 3, hh = blockIdx.y & 7;
  int i0 = blockIdx.x * 32;
  int wave = threadIdx.x >> 5;
  int lane = threadIdx.x & 31;
  int qrow0 = i0 + wave * 16;
  int am = lane & 15, ah = lane >> 4;
  int cn = lane & 15, chalf = lane >> 4;

  // preload q fragments (16 rows x 64, two K-steps of 32)
  ABFrag aq[2];
  {
    const u16* qp = q_bf + ((size_t)(qrow0 + am) * BSZ + b) * (NHEAD * DHEAD) + hh * DHEAD;
    for (int ks = 0; ks < 2; ++ks) {
      const u16* p = qp + ks * 32 + ah * 8;
      aq[ks].q[0] = *(const uint4*)(p);
      aq[ks].q[1] = *(const uint4*)(p + 16);
    }
  }

  v8f o[4], m_run, l_run;
  for (int r = 0; r < 8; ++r) {
    o[0][r] = 0.f; o[1][r] = 0.f; o[2][r] = 0.f; o[3][r] = 0.f;
    m_run[r] = -1e30f; l_run[r] = 0.f;
  }

  int nch = i0 / 32 + 33;                     // chunks covering keys j <= i0+31+MLEN
  for (int ch = 0; ch < nch; ++ch) {
    int j0 = ch * 32;
    // ---- S = (q . k^T) * scale, 16x32 as two 16x16 tiles ----
    v8f s[2];
    for (int t2 = 0; t2 < 2; ++t2) {
      ABFrag bk0, bk1;
      const u16* kp = kv_bf + ((size_t)(j0 + t2 * 16 + cn) * BSZ + b) * (2 * NHEAD * DHEAD)
                      + hh * DHEAD + ah * 8;
      bk0.q[0] = *(const uint4*)(kp);          // d = ah*8 + 0..7, 16 + ah*8 + 0..7
      bk0.q[1] = *(const uint4*)(kp + 16);
      bk1.q[0] = *(const uint4*)(kp + 32);     // d = 32 + (same)
      bk1.q[1] = *(const uint4*)(kp + 48);
      v8f z; for (int r = 0; r < 8; ++r) z[r] = 0.0f;
      z = __builtin_amdgcn_wmma_f32_16x16x32_bf16(false, aq[0].v, false, bk0.v, (short)0, z,
                                                  false, false);
      z = __builtin_amdgcn_wmma_f32_16x16x32_bf16(false, aq[1].v, false, bk1.v, (short)0, z,
                                                  false, false);
      for (int r = 0; r < 8; ++r) s[t2][r] = z[r] * 0.125f;   // 1/sqrt(64)
    }
    // ---- causal+memory mask: key j valid iff j <= i + MLEN ----
    if (j0 + 31 > qrow0 + MLEN) {
      for (int t2 = 0; t2 < 2; ++t2)
        for (int r = 0; r < 8; ++r) {
          int ii = qrow0 + chalf * 8 + r;
          int jj = j0 + t2 * 16 + cn;
          if (jj > ii + MLEN) s[t2][r] = -1e30f;
        }
    }
    // ---- online softmax: row reductions across 16 lanes of each half ----
    v8f mx;
    for (int r = 0; r < 8; ++r) mx[r] = fmaxf(s[0][r], s[1][r]);
    for (int off = 1; off < 16; off <<= 1)
      for (int r = 0; r < 8; ++r) mx[r] = fmaxf(mx[r], __shfl_xor(mx[r], off, 32));
    v8f m_new, alpha, p0, p1, rs;
    for (int r = 0; r < 8; ++r) {
      m_new[r] = fmaxf(m_run[r], mx[r]);
      alpha[r] = __expf(m_run[r] - m_new[r]);
      p0[r] = __expf(s[0][r] - m_new[r]);
      p1[r] = __expf(s[1][r] - m_new[r]);
      rs[r] = p0[r] + p1[r];
    }
    for (int off = 1; off < 16; off <<= 1)
      for (int r = 0; r < 8; ++r) rs[r] += __shfl_xor(rs[r], off, 32);
    for (int r = 0; r < 8; ++r) {
      l_run[r] = l_run[r] * alpha[r] + rs[r];
      m_run[r] = m_new[r];
      o[0][r] *= alpha[r]; o[1][r] *= alpha[r]; o[2][r] *= alpha[r]; o[3][r] *= alpha[r];
    }
    // ---- transpose P through LDS: C-layout -> A-fragment layout ----
    for (int r = 0; r < 8; ++r) {
      pls[wave][chalf * 8 + r][cn]      = f32_to_bf16(p0[r]);
      pls[wave][chalf * 8 + r][16 + cn] = f32_to_bf16(p1[r]);
    }
    __syncthreads();
    ABFrag pa;
    {
      const u16* lp = &pls[wave][am][0];
      pa.q[0] = *(const uint4*)(lp + ah * 8);
      pa.q[1] = *(const uint4*)(lp + 16 + ah * 8);
    }
    __syncthreads();
    // ---- O += P x V  (16x32 . 32x64 as four 16x16 tiles) ----
    const u16* vbase = vT + ((size_t)(b * NHEAD + hh) * DHEAD + cn) * KLEN + j0 + ah * 8;
    for (int c = 0; c < 4; ++c) {
      ABFrag bv;
      const u16* vp = vbase + (size_t)c * 16 * KLEN;
      bv.q[0] = *(const uint4*)(vp);
      bv.q[1] = *(const uint4*)(vp + 16);
      o[c] = __builtin_amdgcn_wmma_f32_16x16x32_bf16(false, pa.v, false, bv.v, (short)0, o[c],
                                                     false, false);
    }
  }
  // ---- finalize: divide by l, store bf16 ----
  v8f inv;
  for (int r = 0; r < 8; ++r) inv[r] = 1.0f / l_run[r];
  for (int c = 0; c < 4; ++c)
    for (int r = 0; r < 8; ++r) {
      int ii = qrow0 + chalf * 8 + r;
      int col = hh * DHEAD + c * 16 + cn;
      vec_bf[((size_t)ii * BSZ + b) * (NHEAD * DHEAD) + col] = f32_to_bf16(o[c][r] * inv[r]);
    }
}

// ---------------- fused residual + LayerNorm ----------------
// grid: 4096 rows; block 256 threads; D=512 -> 2 elems/thread
__global__ void add_ln_kernel(const float* __restrict__ x, const float* __restrict__ y,
                              const float* __restrict__ g, const float* __restrict__ bta,
                              float* __restrict__ outf, u16* __restrict__ outb) {
  __shared__ float rs[256], rss[256];
  int row = blockIdx.x;
  int t = threadIdx.x;
  const float* xr = x + (size_t)row * DMODEL;
  const float* yr = y + (size_t)row * DMODEL;
  float v0 = xr[t] + yr[t];
  float v1 = xr[t + 256] + yr[t + 256];
  rs[t] = v0 + v1;
  rss[t] = v0 * v0 + v1 * v1;
  __syncthreads();
  for (int o2 = 128; o2 > 0; o2 >>= 1) {
    if (t < o2) { rs[t] += rs[t + o2]; rss[t] += rss[t + o2]; }
    __syncthreads();
  }
  float mu = rs[0] * (1.0f / DMODEL);
  float var = rss[0] * (1.0f / DMODEL) - mu * mu;
  float inv = rsqrtf(var + 1e-5f);
  float o0 = (v0 - mu) * inv * g[t] + bta[t];
  float o1 = (v1 - mu) * inv * g[t + 256] + bta[t + 256];
  outf[(size_t)row * DMODEL + t] = o0;
  outf[(size_t)row * DMODEL + t + 256] = o1;
  if (outb) {
    outb[(size_t)row * DMODEL + t] = f32_to_bf16(o0);
    outb[(size_t)row * DMODEL + t + 256] = f32_to_bf16(o1);
  }
}

// ---------------- host orchestration ----------------
extern "C" void kernel_launch(void* const* d_in, const int* in_sizes, int n_in,
                              void* d_out, int out_size, void* d_ws, size_t ws_size,
                              hipStream_t stream) {
  (void)in_sizes; (void)n_in; (void)ws_size;
  const int*   inp  = (const int*)  d_in[0];
  const float* emb  = (const float*)d_in[1];
  const float* mems = (const float*)d_in[2];
  const float* Wq   = (const float*)d_in[3];
  const float* Wkv  = (const float*)d_in[4];
  const float* Wo   = (const float*)d_in[5];
  const float* ln1g = (const float*)d_in[6];
  const float* ln1b = (const float*)d_in[7];
  const float* W1   = (const float*)d_in[8];
  const float* b1   = (const float*)d_in[9];
  const float* W2   = (const float*)d_in[10];
  const float* b2   = (const float*)d_in[11];
  const float* ln2g = (const float*)d_in[12];
  const float* ln2b = (const float*)d_in[13];

  char* wsp = (char*)d_ws;
  auto carve = [&](size_t bytes) { void* p = wsp; wsp += (bytes + 255) & ~(size_t)255; return p; };
  const size_t ROWS = (size_t)QLEN * BSZ;           // 4096
  float* h      = (float*)carve(ROWS * DMODEL * 4);
  float* tmp    = (float*)carve(ROWS * DMODEL * 4); // attn-out / ffn-out
  u16* c_bf   = (u16*)carve((size_t)KLEN * BSZ * DMODEL * 2);
  u16* q_bf   = (u16*)carve(ROWS * DMODEL * 2);
  u16* kv_bf  = (u16*)carve((size_t)KLEN * BSZ * 2 * DMODEL * 2);
  u16* vT     = (u16*)carve((size_t)BSZ * NHEAD * DHEAD * KLEN * 2);
  u16* vec_bf = (u16*)carve(ROWS * DMODEL * 2);
  u16* hb     = (u16*)carve(ROWS * DMODEL * 2);
  u16* ff1_bf = (u16*)carve(ROWS * DFF * 2);
  u16* WqT  = (u16*)carve((size_t)DMODEL * DMODEL * 2);
  u16* WkvT = (u16*)carve((size_t)DMODEL * 2 * DMODEL * 2);
  u16* WoT  = (u16*)carve((size_t)DMODEL * DMODEL * 2);
  u16* W1T  = (u16*)carve((size_t)DMODEL * DFF * 2);
  u16* W2T  = (u16*)carve((size_t)DFF * DMODEL * 2);

  auto gemm = [&](const u16* A, int lda, const u16* Bt, int ldb, float* Cf, u16* Cb, int ldc,
                  const float* bias, int M, int N, int K, int relu) {
    dim3 grid(N / 128, M / 64);
    wmma_gemm_bt<<<grid, 128, 0, stream>>>(A, lda, Bt, ldb, Cf, Cb, ldc, bias, K, relu);
  };

  embed_kernel<<<QLEN * BSZ, 256, 0, stream>>>(inp, emb, h);

  const int HWD = NHEAD * DHEAD;                  // 512
  const int memsz = MLEN * BSZ * DMODEL;          // 2097152 per layer
  for (int l = 0; l < NLAYER; ++l) {
    // weights -> bf16 transposed
    transp_kernel<<<(DMODEL * DMODEL + 255) / 256, 256, 0, stream>>>(Wq + (size_t)l * DMODEL * HWD, WqT, DMODEL, HWD);
    transp_kernel<<<(DMODEL * 2 * HWD + 255) / 256, 256, 0, stream>>>(Wkv + (size_t)l * DMODEL * 2 * HWD, WkvT, DMODEL, 2 * HWD);
    transp_kernel<<<(HWD * DMODEL + 255) / 256, 256, 0, stream>>>(Wo + (size_t)l * HWD * DMODEL, WoT, HWD, DMODEL);
    transp_kernel<<<(DMODEL * DFF + 255) / 256, 256, 0, stream>>>(W1 + (size_t)l * DMODEL * DFF, W1T, DMODEL, DFF);
    transp_kernel<<<(DFF * DMODEL + 255) / 256, 256, 0, stream>>>(W2 + (size_t)l * DFF * DMODEL, W2T, DFF, DMODEL);
    // c = [mems[l]; h] in bf16
    f2bf_kernel<<<memsz / 256, 256, 0, stream>>>(mems + (size_t)l * memsz, c_bf, memsz);
    f2bf_kernel<<<memsz / 256, 256, 0, stream>>>(h, c_bf + memsz, memsz);
    // q = h @ Wq ; kv = c @ Wkv   (bf16 outputs, f32 accumulation)
    gemm(c_bf + memsz, DMODEL, WqT, DMODEL, nullptr, q_bf, HWD, nullptr, QLEN * BSZ, HWD, DMODEL, 0);
    gemm(c_bf, DMODEL, WkvT, DMODEL, nullptr, kv_bf, 2 * HWD, nullptr, KLEN * BSZ, 2 * HWD, DMODEL, 0);
    // v -> [b,h,d,klen] for contiguous P.V fragments
    vtrans_kernel<<<(BSZ * NHEAD * DHEAD * KLEN) / 256, 256, 0, stream>>>(kv_bf, vT);
    // flash attention
    flash_attn<<<dim3(QLEN / 32, BSZ * NHEAD), 64, 0, stream>>>(q_bf, kv_bf, vT, vec_bf);
    // out-proj, residual + LN1 (writes f32 h and bf16 hb)
    gemm(vec_bf, HWD, WoT, HWD, tmp, nullptr, DMODEL, nullptr, QLEN * BSZ, DMODEL, HWD, 0);
    add_ln_kernel<<<QLEN * BSZ, 256, 0, stream>>>(h, tmp, ln1g + (size_t)l * DMODEL, ln1b + (size_t)l * DMODEL, h, hb);
    // FFN
    gemm(hb, DMODEL, W1T, DMODEL, nullptr, ff1_bf, DFF, b1 + (size_t)l * DFF, QLEN * BSZ, DFF, DMODEL, 1);
    gemm(ff1_bf, DFF, W2T, DFF, tmp, nullptr, DMODEL, b2 + (size_t)l * DMODEL, QLEN * BSZ, DMODEL, DFF, 0);
    add_ln_kernel<<<QLEN * BSZ, 256, 0, stream>>>(h, tmp, ln2g + (size_t)l * DMODEL, ln2b + (size_t)l * DMODEL, h, nullptr);
  }
  hipMemcpyAsync(d_out, h, (size_t)out_size * sizeof(float), hipMemcpyDeviceToDevice, stream);
}
